// SwinTransformer_26508538151574
// MI455X (gfx1250) — matrix operand order repaired and checked by reading, not compiled, not run on hardware
//
#include <hip/hip_runtime.h>
#include <hip/hip_bf16.h>
#include <math.h>

typedef _Float16 v16h __attribute__((ext_vector_type(16)));
typedef _Float16 v8h  __attribute__((ext_vector_type(8)));
typedef float    v8f  __attribute__((ext_vector_type(8)));

#define DIMC   256
#define NHEADS 8
#define HD     32
#define WH     5
#define WW     2
#define NTOK   10      // WH*WW
#define HWIN   16      // 80/5
#define WWIN   32      // 64/2
#define LTOK   5120    // 80*64
#define BATCH  32
#define NROWS  (BATCH*LTOK)     // 163840 token rows
#define MROWS  64               // token rows per MLP block
#define H2_BYTES ((size_t)NROWS * DIMC * sizeof(_Float16))   // 83,886,080
#define WT_BYTES ((size_t)DIMC * DIMC * sizeof(_Float16))    // 131,072

// Build v16h in the CDNA5 A-operand half order from a row pointer at k0:
// halves 0-7 -> K = k0+sel8+0..7 ; halves 8-15 -> K = k0+16+sel8+0..7
__device__ __forceinline__ v16h ldA(const _Float16* p, int sel8) {
    v8h a0 = *(const v8h*)(p + sel8);
    v8h a1 = *(const v8h*)(p + 16 + sel8);
    return __builtin_shufflevector(a0, a1, 0,1,2,3,4,5,6,7,8,9,10,11,12,13,14,15);
}
// B-operand: same half order as A (p already includes +sel8): K = sel8+0..7, 16+sel8+0..7
__device__ __forceinline__ v16h ldB(const _Float16* p) {
    v8h b0 = *(const v8h*)p;
    v8h b1 = *(const v8h*)(p + 16);
    return __builtin_shufflevector(b0, b1, 0,1,2,3,4,5,6,7,8,9,10,11,12,13,14,15);
}

// ---------------------------------------------------------------------------
// Kernel 0: transpose + f32->f16 convert MLP weights (w[k][n] -> wt[n][k])
// ---------------------------------------------------------------------------
__global__ __launch_bounds__(256) void swin_convert_weights(
    const float* __restrict__ w1, const float* __restrict__ w2,
    _Float16* __restrict__ w1t, _Float16* __restrict__ w2t)
{
    int idx = blockIdx.x * 256 + threadIdx.x;     // 0 .. 65535
    int k = idx >> 8;
    int n = idx & 255;
    w1t[n * DIMC + k] = (_Float16)w1[idx];
    w2t[n * DIMC + k] = (_Float16)w2[idx];
}

// ---------------------------------------------------------------------------
// Kernel 1: per-window LN1 -> attention (q=k=v quirk) -> residual -> LN2
//   writes: y (residual, f32) into d_out ; h2 = LN2(y) as f16 into workspace
// one block (256 thr = 8 waves) per window; 16384 windows; x read ONCE (LDS)
// ---------------------------------------------------------------------------
__global__ __launch_bounds__(256) void swin_attn_ln2(
    const float* __restrict__ x,
    const float* __restrict__ g1, const float* __restrict__ b1,
    const float* __restrict__ bias_tab,
    const float* __restrict__ g2, const float* __restrict__ b2,
    float* __restrict__ y_out, _Float16* __restrict__ h2)
{
    __shared__ float xbuf[NTOK][DIMC];              // raw window input
    __shared__ float hbuf[NTOK][DIMC];              // LN1 output (q=k=v)
    __shared__ float scores[NHEADS][NTOK][NTOK];

    const int wid  = blockIdx.x;
    const int b    = wid / (HWIN * WWIN);
    const int rem  = wid % (HWIN * WWIN);
    const int wi   = rem / WWIN;
    const int wj   = rem % WWIN;
    const int tid  = threadIdx.x;
    const int lane = tid & 31;
    const int wave = tid >> 5;

    // ---- LN1: one wave per token (also stashes x into LDS) ----
    for (int n = wave; n < NTOK; n += 8) {
        const int r = n >> 1, c = n & 1;
        const size_t gb = ((size_t)b * LTOK + (size_t)(wi * WH + r) * 64 + (wj * WW + c)) * DIMC;
        float v[8], s = 0.f, ss = 0.f;
        #pragma unroll
        for (int k = 0; k < 8; ++k) {
            float val = x[gb + lane + 32 * k];
            xbuf[n][lane + 32 * k] = val;
            v[k] = val; s += val; ss += val * val;
        }
        #pragma unroll
        for (int off = 16; off >= 1; off >>= 1) {
            s  += __shfl_xor(s,  off, 32);
            ss += __shfl_xor(ss, off, 32);
        }
        const float mean = s * (1.f / DIMC);
        const float var  = ss * (1.f / DIMC) - mean * mean;
        const float rstd = rsqrtf(var + 1e-5f);
        #pragma unroll
        for (int k = 0; k < 8; ++k) {
            int ch = lane + 32 * k;
            hbuf[n][ch] = (v[k] - mean) * rstd * g1[ch] + b1[ch];
        }
    }
    __syncthreads();

    // ---- scores: (q*scale) . k + relative position bias ----
    const float scale = 0.17677669529663687f;   // 1/sqrt(32)
    for (int idx = tid; idx < NHEADS * NTOK * NTOK; idx += 256) {
        const int h = idx / (NTOK * NTOK);
        const int q = (idx / NTOK) % NTOK;
        const int k = idx % NTOK;
        float d = 0.f;
        #pragma unroll
        for (int c = 0; c < HD; ++c)
            d += hbuf[q][h * HD + c] * hbuf[k][h * HD + c];
        const int qr = q >> 1, qc = q & 1, kr = k >> 1, kc = k & 1;
        const int rel = (qr - kr + (WH - 1)) * (2 * WW - 1) + (qc - kc + (WW - 1));
        scores[h][q][k] = d * scale + bias_tab[rel * NHEADS + h];
    }
    __syncthreads();

    // ---- softmax along k ----
    for (int rowi = tid; rowi < NHEADS * NTOK; rowi += 256) {
        const int h = rowi / NTOK, q = rowi % NTOK;
        float mx = -1e30f;
        #pragma unroll
        for (int k = 0; k < NTOK; ++k) mx = fmaxf(mx, scores[h][q][k]);
        float e[NTOK], sum = 0.f;
        #pragma unroll
        for (int k = 0; k < NTOK; ++k) { e[k] = __expf(scores[h][q][k] - mx); sum += e[k]; }
        const float inv = 1.f / sum;
        #pragma unroll
        for (int k = 0; k < NTOK; ++k) scores[h][q][k] = e[k] * inv;
    }
    __syncthreads();

    // ---- out = attn @ v ; residual ; fused LN2 ; one wave per token ----
    for (int n = wave; n < NTOK; n += 8) {
        const int r = n >> 1, c = n & 1;
        const size_t gb = ((size_t)b * LTOK + (size_t)(wi * WH + r) * 64 + (wj * WW + c)) * DIMC;
        float yv[8], s = 0.f, ss = 0.f;
        #pragma unroll
        for (int kk = 0; kk < 8; ++kk) {
            const int ch = lane + 32 * kk;
            const int h  = ch >> 5;           // HD == 32
            float o = 0.f;
            #pragma unroll
            for (int k = 0; k < NTOK; ++k)
                o += scores[h][n][k] * hbuf[k][ch];
            const float val = xbuf[n][ch] + o;
            yv[kk] = val; s += val; ss += val * val;
        }
        #pragma unroll
        for (int off = 16; off >= 1; off >>= 1) {
            s  += __shfl_xor(s,  off, 32);
            ss += __shfl_xor(ss, off, 32);
        }
        const float mean = s * (1.f / DIMC);
        const float var  = ss * (1.f / DIMC) - mean * mean;
        const float rstd = rsqrtf(var + 1e-5f);
        #pragma unroll
        for (int kk = 0; kk < 8; ++kk) {
            const int ch = lane + 32 * kk;
            y_out[gb + ch] = yv[kk];
            h2[gb + ch] = (_Float16)((yv[kk] - mean) * rstd * g2[ch] + b2[ch]);
        }
    }
}

// ---------------------------------------------------------------------------
// Kernel 2: fused MLP via WMMA f16 (fp32 accum)
//   out = y + gelu(h2 @ W1 + b1) @ W2 + b2     (y already resides in d_out)
// block: 256 thr = 8 waves (2 M-groups x 4 N-slabs); 64 rows x 256 cols/block
// each wave owns TWO 16-row M-tiles -> every B-tile register load feeds 2 WMMAs
// ---------------------------------------------------------------------------
__global__ __launch_bounds__(256) void swin_mlp_wmma(
    const _Float16* __restrict__ h2,
    const _Float16* __restrict__ w1t, const float* __restrict__ b1,
    const _Float16* __restrict__ w2t, const float* __restrict__ b2,
    float* __restrict__ out)
{
    __shared__ __align__(16) _Float16 inter[MROWS][DIMC];   // 32 KB gelu intermediate

    const int tid  = threadIdx.x;
    const int lane = tid & 31;
    const int wave = tid >> 5;
    const int wm   = wave >> 2;         // 0..1 : M group
    const int wn   = wave & 3;          // 0..3 : 64-col slab
    const int lmod = lane & 15;
    const int lhi  = lane >> 4;         // 0 | 1
    const int sel8  = lhi * 8;
    const int sel16 = lhi * 16;
    const size_t row0 = (size_t)blockIdx.x * MROWS;
    const int mloc0 = wm * 16;          // this wave's two m-tiles
    const int mloc1 = wm * 16 + 32;

    // ================= GEMM1: h2 @ W1 =================
    v8f acc[4][2];
    #pragma unroll
    for (int nt = 0; nt < 4; ++nt)
        #pragma unroll
        for (int t = 0; t < 2; ++t)
            #pragma unroll
            for (int r = 0; r < 8; ++r) acc[nt][t][r] = 0.f;

    const _Float16* arow0 = h2 + (row0 + mloc0 + lmod) * DIMC;
    const _Float16* arow1 = h2 + (row0 + mloc1 + lmod) * DIMC;
    for (int k0 = 0; k0 < DIMC; k0 += 32) {
        v16h a0 = ldA(arow0 + k0, sel8);
        v16h a1 = ldA(arow1 + k0, sel8);
        #pragma unroll
        for (int nt = 0; nt < 4; ++nt) {
            const int ncol = wn * 64 + nt * 16 + lmod;
            const _Float16* bp = w1t + (size_t)ncol * DIMC + k0 + sel8;
            __builtin_prefetch(bp + 32, 0, 3);  // next k-step, WGP-scope prefetch
            v16h bm = ldB(bp);
            acc[nt][0] = __builtin_amdgcn_wmma_f32_16x16x32_f16(
                false, a0, false, bm, (short)0, acc[nt][0], false, false);
            acc[nt][1] = __builtin_amdgcn_wmma_f32_16x16x32_f16(
                false, a1, false, bm, (short)0, acc[nt][1], false, false);
        }
    }

    // epilogue 1: +b1, exact GELU, f16 -> LDS   (D: lane=col l&15, vgpr r -> m=r+lhi*8)
    #pragma unroll
    for (int nt = 0; nt < 4; ++nt) {
        const int ncol = wn * 64 + nt * 16 + lmod;
        const float bias = b1[ncol];
        #pragma unroll
        for (int t = 0; t < 2; ++t) {
            const int mb = wm * 16 + t * 32;
            #pragma unroll
            for (int r = 0; r < 8; ++r) {
                const int m = mb + r + lhi * 8;
                const float v = acc[nt][t][r] + bias;
                const float g = 0.5f * v * (1.f + erff(v * 0.70710678118654752f));
                inter[m][ncol] = (_Float16)g;
            }
        }
    }
    __syncthreads();

    // ================= GEMM2: gelu(.) @ W2 =================
    v8f acc2[4][2];
    #pragma unroll
    for (int nt = 0; nt < 4; ++nt)
        #pragma unroll
        for (int t = 0; t < 2; ++t)
            #pragma unroll
            for (int r = 0; r < 8; ++r) acc2[nt][t][r] = 0.f;

    const _Float16* irow0 = &inter[mloc0 + lmod][0];
    const _Float16* irow1 = &inter[mloc1 + lmod][0];
    for (int k0 = 0; k0 < DIMC; k0 += 32) {
        v16h a0 = ldA(irow0 + k0, sel8);
        v16h a1 = ldA(irow1 + k0, sel8);
        #pragma unroll
        for (int nt = 0; nt < 4; ++nt) {
            const int ncol = wn * 64 + nt * 16 + lmod;
            const _Float16* bp = w2t + (size_t)ncol * DIMC + k0 + sel8;
            __builtin_prefetch(bp + 32, 0, 3);
            v16h bm = ldB(bp);
            acc2[nt][0] = __builtin_amdgcn_wmma_f32_16x16x32_f16(
                false, a0, false, bm, (short)0, acc2[nt][0], false, false);
            acc2[nt][1] = __builtin_amdgcn_wmma_f32_16x16x32_f16(
                false, a1, false, bm, (short)0, acc2[nt][1], false, false);
        }
    }

    // epilogue 2: + b2 + residual y (already in out), final store
    #pragma unroll
    for (int nt = 0; nt < 4; ++nt) {
        const int ncol = wn * 64 + nt * 16 + lmod;
        const float bias = b2[ncol];
        #pragma unroll
        for (int t = 0; t < 2; ++t) {
            const int mb = wm * 16 + t * 32;
            #pragma unroll
            for (int r = 0; r < 8; ++r) {
                const int m = mb + r + lhi * 8;
                const size_t gi = (row0 + m) * DIMC + ncol;
                out[gi] = out[gi] + bias + acc2[nt][t][r];
            }
        }
    }
}

// ---------------------------------------------------------------------------
extern "C" void kernel_launch(void* const* d_in, const int* in_sizes, int n_in,
                              void* d_out, int out_size, void* d_ws, size_t ws_size,
                              hipStream_t stream) {
    const float* x        = (const float*)d_in[0];
    const float* norm1_g  = (const float*)d_in[1];
    const float* norm1_b  = (const float*)d_in[2];
    const float* bias_tab = (const float*)d_in[3];
    // d_in[4], d_in[5]: proj_w / proj_b — faithfully unused (source quirk)
    const float* norm2_g  = (const float*)d_in[6];
    const float* norm2_b  = (const float*)d_in[7];
    const float* mlp1_w   = (const float*)d_in[8];
    const float* mlp1_b   = (const float*)d_in[9];
    const float* mlp2_w   = (const float*)d_in[10];
    const float* mlp2_b   = (const float*)d_in[11];
    float* out = (float*)d_out;

    _Float16* h2  = (_Float16*)d_ws;
    _Float16* w1t = (_Float16*)((char*)d_ws + H2_BYTES);
    _Float16* w2t = (_Float16*)((char*)d_ws + H2_BYTES + WT_BYTES);

    swin_convert_weights<<<DIMC, 256, 0, stream>>>(mlp1_w, mlp2_w, w1t, w2t);

    swin_attn_ln2<<<BATCH * HWIN * WWIN, 256, 0, stream>>>(
        x, norm1_g, norm1_b, bias_tab, norm2_g, norm2_b, out, h2);

    swin_mlp_wmma<<<NROWS / MROWS, 256, 0, stream>>>(
        h2, w1t, mlp1_b, w2t, mlp2_b, out);
}
